// Graph_U_9345848836803
// MI455X (gfx1250) — matrix operand order repaired and checked by reading, not compiled, not run on hardware
//
#include <hip/hip_runtime.h>
#include <hip/hip_bf16.h>

// ---------------------------------------------------------------------------
// Graph U-Net (MinCutPool) inference on MI455X (gfx1250).
// Heavy GEMMs run on v_wmma_f32_16x16x32_bf16 (wave32, f32 accumulate).
// au = S Ap S^T is never materialized (factorized); up-blocks 0/1 are dead.
// ---------------------------------------------------------------------------

typedef __bf16 bf16;
typedef __attribute__((ext_vector_type(16))) __bf16 v16bf;
typedef __attribute__((ext_vector_type(8)))  __bf16 v8bf;
typedef __attribute__((ext_vector_type(8)))  float  v8f;

#define NNODES 6000
#define NFEAT  6
#define NCLS   20
#define NFILT  32

#define LDSPAD 40   // 80B row stride: 16B-aligned for b128, 20-dword conflict-free

// ---------------------------------------------------------------------------
// Tiled bf16 WMMA GEMM:  C[M,N] (f32) = A[M,K] * B[K,N]  (+bias)(+relu)
// Block: WR*WC waves. Wave tile: 32 x (16*SN)  (2 x SN wmma accumulators).
// Block tile: (WR*32) x (WC*16*SN), K-step 32.
// B tile is staged TRANSPOSED in LDS so both fragments are K-contiguous
// 16B runs -> ds_load_b128, zero b16 packing.
// epi: 0=none, 1=+bias, 2=+bias,relu
// ---------------------------------------------------------------------------
template <int WR, int WC, int SN, bool AF32>
__global__ void __launch_bounds__(WR * WC * 32)
gemm_wmma(const void* __restrict__ Ag, const bf16* __restrict__ Bg,
          float* __restrict__ Cg, const float* __restrict__ bias,
          int M, int N, int K, int epi) {
  constexpr int BM = WR * 32;
  constexpr int BN = WC * 16 * SN;
  constexpr int BK = 32;
  constexpr int NT = WR * WC * 32;
  constexpr int AV = (BM * BK) / (8 * NT);   // v8bf chunks per thread (A)
  constexpr int BV = (BK * BN) / (8 * NT);   // v8bf chunks per thread (B)

  __shared__ alignas(16) bf16 As[BM][LDSPAD];   // [m][k]
  __shared__ alignas(16) bf16 Bs[BN][LDSPAD];   // [n][k]  (transposed!)

  const int tid  = threadIdx.x;
  const int wave = tid >> 5;
  const int lane = tid & 31;
  const int wm   = (wave / WC) * 32;        // wave row offset in block tile
  const int wn   = (wave % WC) * 16 * SN;   // wave col offset in block tile
  const int hh   = lane >> 4;               // half-wave select
  const int lc   = lane & 15;

  const int bm0 = blockIdx.y * BM;
  const int bn0 = blockIdx.x * BN;

  v8f acc[2][SN];
#pragma unroll
  for (int i = 0; i < 2; ++i)
#pragma unroll
    for (int j = 0; j < SN; ++j) acc[i][j] = (v8f){};

  for (int k0 = 0; k0 < K; k0 += BK) {
    const bool fullK = (k0 + BK <= K);
    // ---------------- stage A tile: As[m][k] ----------------
    if (fullK && (bm0 + BM <= M) && ((K & 7) == 0)) {
#pragma unroll
      for (int v = 0; v < AV; ++v) {
        const int lin = (tid + v * NT) * 8;
        const int r = lin / BK, c = lin & (BK - 1);
        if (AF32) {
          const float* p = (const float*)Ag + (size_t)(bm0 + r) * K + k0 + c;
          v8bf d;
#pragma unroll
          for (int j = 0; j < 8; ++j) d[j] = (bf16)p[j];
          *(v8bf*)&As[r][c] = d;
        } else {
          *(v8bf*)&As[r][c] =
              *(const v8bf*)((const bf16*)Ag + (size_t)(bm0 + r) * K + k0 + c);
        }
      }
    } else {
      for (int idx = tid; idx < BM * BK; idx += NT) {
        const int r = idx / BK, c = idx & (BK - 1);
        const int gr = bm0 + r, gc = k0 + c;
        bf16 v = (bf16)0.0f;
        if (gr < M && gc < K) {
          if (AF32) v = (bf16)(((const float*)Ag)[(size_t)gr * K + gc]);
          else      v = ((const bf16*)Ag)[(size_t)gr * K + gc];
        }
        As[r][c] = v;
      }
    }
    // ---------------- stage B tile transposed: Bs[n][k] ----------------
    if (fullK && (bn0 + BN <= N) && ((N & 7) == 0)) {
#pragma unroll
      for (int v = 0; v < BV; ++v) {
        const int lin = (tid + v * NT) * 8;
        const int r = lin / BN, c = lin & (BN - 1);
        const v8bf d = *(const v8bf*)(Bg + (size_t)(k0 + r) * N + bn0 + c);
#pragma unroll
        for (int j = 0; j < 8; ++j) Bs[c + j][r] = d[j];
      }
    } else {
      for (int idx = tid; idx < BK * BN; idx += NT) {
        const int r = idx / BN, c = idx & (BN - 1);
        const int gr = k0 + r, gc = bn0 + c;
        bf16 v = (bf16)0.0f;
        if (gr < K && gc < N) v = Bg[(size_t)gr * N + gc];
        Bs[c][r] = v;
      }
    }
    // prefetch next K-tile into cache while this one is consumed
    if (k0 + BK < K) {
      const int pr = bm0 + (tid % BM);
      if (pr < M) {
        if (AF32) __builtin_prefetch(&((const float*)Ag)[(size_t)pr * K + k0 + BK], 0, 0);
        else      __builtin_prefetch(&((const bf16*)Ag)[(size_t)pr * K + k0 + BK], 0, 0);
      }
      const int pc = bn0 + (tid % BN);
      if (pc < N) __builtin_prefetch(&Bg[(size_t)(k0 + BK) * N + pc], 0, 0);
    }
    __syncthreads();

    // ---- fragments: each half is one aligned 16B LDS read (ds_load_b128) ----
    // A 16x32 layout: lanes0-15 K=0..7,16..23; lanes16-31 K=8..15,24..31
    // B 32x16 layout: lanes0-15 K=0..7,16..23; lanes16-31 K=8..15,24..31
    v16bf afr[2], bfr[SN];
#pragma unroll
    for (int s = 0; s < 2; ++s) {
      const v8bf alo = *(const v8bf*)&As[wm + s * 16 + lc][8 * hh];
      const v8bf ahi = *(const v8bf*)&As[wm + s * 16 + lc][16 + 8 * hh];
      afr[s] = __builtin_shufflevector(alo, ahi, 0, 1, 2, 3, 4, 5, 6, 7,
                                       8, 9, 10, 11, 12, 13, 14, 15);
    }
#pragma unroll
    for (int s = 0; s < SN; ++s) {
      const v8bf blo = *(const v8bf*)&Bs[wn + s * 16 + lc][8 * hh];
      const v8bf bhi = *(const v8bf*)&Bs[wn + s * 16 + lc][16 + 8 * hh];
      bfr[s] = __builtin_shufflevector(blo, bhi, 0, 1, 2, 3, 4, 5, 6, 7,
                                       8, 9, 10, 11, 12, 13, 14, 15);
    }
#pragma unroll
    for (int i = 0; i < 2; ++i)
#pragma unroll
      for (int j = 0; j < SN; ++j)
        acc[i][j] = __builtin_amdgcn_wmma_f32_16x16x32_bf16(
            false, afr[i], false, bfr[j], (short)0, acc[i][j], false, false);
    __syncthreads();
  }

  // ---- epilogue: C layout VGPR r -> row r (+8 upper half), col lane&15 ----
#pragma unroll
  for (int i = 0; i < 2; ++i)
#pragma unroll
    for (int j = 0; j < SN; ++j)
#pragma unroll
      for (int r = 0; r < 8; ++r) {
        const int row = bm0 + wm + i * 16 + r + 8 * hh;
        const int col = bn0 + wn + j * 16 + lc;
        if (row < M && col < N) {
          float v = acc[i][j][r];
          if (epi >= 1) v += bias[col];
          if (epi == 2) v = v > 0.0f ? v : 0.0f;
          Cg[(size_t)row * N + col] = v;
        }
      }
}

// ---------------------------------------------------------------------------
// Elementwise / reduction helpers
// ---------------------------------------------------------------------------
__global__ void cvt_bf16(const float* __restrict__ x, bf16* __restrict__ y, int n) {
  for (int i = blockIdx.x * blockDim.x + threadIdx.x; i < n; i += gridDim.x * blockDim.x)
    y[i] = (bf16)x[i];
}

// write both row-major bf16 copy and transposed bf16 copy of s [rows,cols]
__global__ void cvt_dual(const float* __restrict__ s, bf16* __restrict__ Sb,
                         bf16* __restrict__ STb, int rows, int cols) {
  const size_t tot = (size_t)rows * cols;
  for (size_t i = blockIdx.x * (size_t)blockDim.x + threadIdx.x; i < tot;
       i += (size_t)gridDim.x * blockDim.x) {
    const int r = (int)(i / cols), c = (int)(i % cols);
    const bf16 v = (bf16)s[i];
    Sb[i] = v;
    STb[(size_t)c * rows + r] = v;
  }
}

__global__ void __launch_bounds__(256) softmax_rows(float* __restrict__ x, int cols) {
  const int r = blockIdx.x;
  float* row = x + (size_t)r * cols;
  __shared__ float red[256];
  float mx = -3.0e38f;
  for (int c = threadIdx.x; c < cols; c += 256) mx = fmaxf(mx, row[c]);
  red[threadIdx.x] = mx; __syncthreads();
  for (int s = 128; s > 0; s >>= 1) {
    if (threadIdx.x < s) red[threadIdx.x] = fmaxf(red[threadIdx.x], red[threadIdx.x + s]);
    __syncthreads();
  }
  mx = red[0]; __syncthreads();
  float sm = 0.0f;
  for (int c = threadIdx.x; c < cols; c += 256) {
    const float e = __expf(row[c] - mx);
    row[c] = e; sm += e;
  }
  red[threadIdx.x] = sm; __syncthreads();
  for (int s = 128; s > 0; s >>= 1) {
    if (threadIdx.x < s) red[threadIdx.x] += red[threadIdx.x + s];
    __syncthreads();
  }
  const float inv = 1.0f / red[0];
  for (int c = threadIdx.x; c < cols; c += 256) row[c] *= inv;
}

// norm_adj pass 1: degree per row (diagonal excluded), d = deg>0 ? rsqrt : 0
__global__ void __launch_bounds__(256) rowdeg(const float* __restrict__ Mt,
                                              float* __restrict__ d, int k) {
  const int r = blockIdx.x;
  const float* row = Mt + (size_t)r * k;
  __shared__ float red[256];
  float sm = 0.0f;
  for (int c = threadIdx.x; c < k; c += 256)
    if (c != r) sm += row[c];
  red[threadIdx.x] = sm; __syncthreads();
  for (int s = 128; s > 0; s >>= 1) {
    if (threadIdx.x < s) red[threadIdx.x] += red[threadIdx.x + s];
    __syncthreads();
  }
  if (threadIdx.x == 0) d[r] = (red[0] > 0.0f) ? rsqrtf(red[0]) : 0.0f;
}

// norm_adj pass 2: Apb = bf16( (i==j) ? 0 : M[i][j]*d[i]*d[j] )
__global__ void normadj_scale(const float* __restrict__ Mt, const float* __restrict__ d,
                              bf16* __restrict__ Apb, int k) {
  const size_t tot = (size_t)k * k;
  for (size_t i = blockIdx.x * (size_t)blockDim.x + threadIdx.x; i < tot;
       i += (size_t)gridDim.x * blockDim.x) {
    const int r = (int)(i / k), c = (int)(i % k);
    const float v = (r == c) ? 0.0f : Mt[i] * d[r] * d[c];
    Apb[i] = (bf16)v;
  }
}

// per-column (cols=32) training-mode BatchNorm; emits f32 + bf16
__global__ void __launch_bounds__(256) bn_cols(const float* __restrict__ x,
                                               const float* __restrict__ g,
                                               const float* __restrict__ b,
                                               float* __restrict__ yf,
                                               bf16* __restrict__ yb,
                                               int rows, int cols) {
  const int c = blockIdx.x;
  __shared__ float r1[256], r2[256];
  float sm = 0.0f, sq = 0.0f;
  for (int r = threadIdx.x; r < rows; r += 256) {
    const float v = x[(size_t)r * cols + c];
    sm += v; sq += v * v;
  }
  r1[threadIdx.x] = sm; r2[threadIdx.x] = sq; __syncthreads();
  for (int s = 128; s > 0; s >>= 1) {
    if (threadIdx.x < s) { r1[threadIdx.x] += r1[threadIdx.x + s]; r2[threadIdx.x] += r2[threadIdx.x + s]; }
    __syncthreads();
  }
  const float mean = r1[0] / rows;
  const float var  = r2[0] / rows - mean * mean;
  const float sc = g[c] * rsqrtf(var + 1e-3f);
  const float sh = b[c] - mean * sc;
  for (int r = threadIdx.x; r < rows; r += 256) {
    const float v = x[(size_t)r * cols + c] * sc + sh;
    yf[(size_t)r * cols + c] = v;
    yb[(size_t)r * cols + c] = (bf16)v;
  }
}

// ---------------------------------------------------------------------------
// Host orchestration
// ---------------------------------------------------------------------------
extern "C" void kernel_launch(void* const* d_in, const int* in_sizes, int n_in,
                              void* d_out, int out_size, void* d_ws, size_t ws_size,
                              hipStream_t stream) {
  (void)in_sizes; (void)n_in; (void)out_size; (void)ws_size;
  const int KS[3] = {3000, 1500, 750};

  // --- inputs (setup_inputs dict order, params flattened depth-first) ---
  const float* X = (const float*)d_in[0];
  const float* A = (const float*)d_in[1];
  const float* Ws[3]; const float* bsv[3];
  for (int i = 0; i < 3; ++i) { Ws[i] = (const float*)d_in[2 + i]; bsv[i] = (const float*)d_in[5 + i]; }
  const float *Wg[6], *bg[6], *gam[6], *bet[6];
  for (int i = 0; i < 6; ++i) {
    Wg[i]  = (const float*)d_in[8 + i];
    bg[i]  = (const float*)d_in[14 + i];
    gam[i] = (const float*)d_in[20 + i];
    bet[i] = (const float*)d_in[26 + i];
  }
  const float* Wout = (const float*)d_in[32];
  const float* bout = (const float*)d_in[33];

  // --- workspace carving ---
  char* wp = (char*)d_ws;
  auto carve = [&](size_t bytes) -> char* {
    char* p = wp; wp += (bytes + 255) & ~(size_t)255; return p;
  };
  bf16* Xb = (bf16*)carve(sizeof(bf16) * NNODES * NFEAT);
  bf16* Wsb[3]; bf16* Sb[3]; bf16* STb[3]; bf16* Apb[3]; bf16* xab[3];
  for (int i = 0; i < 3; ++i) {
    Wsb[i] = (bf16*)carve(sizeof(bf16) * NFEAT * KS[i]);
    Sb[i]  = (bf16*)carve(sizeof(bf16) * (size_t)NNODES * KS[i]);
    STb[i] = (bf16*)carve(sizeof(bf16) * (size_t)KS[i] * NNODES);
    Apb[i] = (bf16*)carve(sizeof(bf16) * (size_t)KS[i] * KS[i]);
    xab[i] = (bf16*)carve(sizeof(bf16) * KS[i] * NFILT);
  }
  bf16* Wgb[6];
  for (int i = 0; i < 6; ++i)
    Wgb[i] = (bf16*)carve(sizeof(bf16) * ((i < 3) ? NFEAT : NFILT) * NFILT);
  bf16* Woutb = (bf16*)carve(sizeof(bf16) * NFILT * NCLS);
  float* S1   = (float*)carve(sizeof(float) * (size_t)NNODES * 3000);  // shared big f32 scratch
  bf16*  Bb   = (bf16*)carve(sizeof(bf16) * (size_t)NNODES * 3000);    // A@s in bf16
  float* dvec = (float*)carve(sizeof(float) * 3000);
  float* xpf  = (float*)carve(sizeof(float) * 3000 * NFEAT);
  bf16*  xpb  = (bf16*)carve(sizeof(bf16) * 3000 * NFEAT);
  float* zf   = (float*)carve(sizeof(float) * NNODES * NFILT);
  bf16*  zb   = (bf16*)carve(sizeof(bf16) * NNODES * NFILT);
  float* hf   = (float*)carve(sizeof(float) * NNODES * NFILT);
  float* x5f  = (float*)carve(sizeof(float) * NNODES * NFILT);
  bf16*  x5b  = (bf16*)carve(sizeof(bf16) * NNODES * NFILT);

  auto cvt = [&](const float* src, bf16* dst, int n) {
    cvt_bf16<<<dim3((n + 255) / 256), dim3(256), 0, stream>>>(src, dst, n);
  };
  // big GEMM: 64x256 block tiles, 8 waves, 32x64 per wave (8 wmma / K-step)
  auto gemmBig = [&](const void* Ain, const bf16* Bin, float* C, const float* bias,
                     int M, int N, int K, int epi, bool af32) {
    dim3 g((N + 255) / 256, (M + 63) / 64);
    if (af32) gemm_wmma<2, 4, 4, true ><<<g, dim3(256), 0, stream>>>(Ain, Bin, C, bias, M, N, K, epi);
    else      gemm_wmma<2, 4, 4, false><<<g, dim3(256), 0, stream>>>(Ain, Bin, C, bias, M, N, K, epi);
  };
  // skinny-N GEMM: 128x32 block tiles, 4 waves, 32x32 per wave
  auto gemmTall = [&](const bf16* Ain, const bf16* Bin, float* C, const float* bias,
                      int M, int N, int K, int epi) {
    dim3 g((N + 31) / 32, (M + 127) / 128);
    gemm_wmma<4, 1, 2, false><<<g, dim3(128), 0, stream>>>((const void*)Ain, Bin, C, bias, M, N, K, epi);
  };

  // --- weight conversion ---
  cvt(X, Xb, NNODES * NFEAT);
  for (int i = 0; i < 3; ++i) cvt(Ws[i], Wsb[i], NFEAT * KS[i]);
  for (int i = 0; i < 6; ++i) cvt(Wg[i], Wgb[i], ((i < 3) ? NFEAT : NFILT) * NFILT);
  cvt(Wout, Woutb, NFILT * NCLS);

  // --- down path: 3 pooling levels ---
  for (int i = 0; i < 3; ++i) {
    const int k = KS[i];
    // s = softmax(X @ Ws + bs)   [N,k]
    gemmBig(Xb, Wsb[i], S1, bsv[i], NNODES, k, NFEAT, 1, false);
    softmax_rows<<<dim3(NNODES), dim3(256), 0, stream>>>(S1, k);
    {
      const size_t tot = (size_t)NNODES * k;
      cvt_dual<<<dim3((unsigned)((tot + 255) / 256 > 16384 ? 16384 : (tot + 255) / 256)),
                 dim3(256), 0, stream>>>(S1, Sb[i], STb[i], NNODES, k);
    }
    // B = A @ s   [N,k]  (A read as f32, converted in the LDS stage)
    gemmBig(A, Sb[i], S1, nullptr, NNODES, k, NNODES, 0, true);
    cvt(S1, Bb, NNODES * k);
    // M = s^T @ B  [k,k]  -> norm_adj -> Apb (bf16)
    gemmBig(STb[i], Bb, S1, nullptr, k, k, NNODES, 0, false);
    rowdeg<<<dim3(k), dim3(256), 0, stream>>>(S1, dvec, k);
    {
      const size_t tot = (size_t)k * k;
      normadj_scale<<<dim3((unsigned)((tot + 255) / 256 > 16384 ? 16384 : (tot + 255) / 256)),
                      dim3(256), 0, stream>>>(S1, dvec, Apb[i], k);
    }
    // xp = s^T @ X  [k,F]
    gemmTall(STb[i], Xb, xpf, nullptr, k, NFEAT, NNODES, 0);
    cvt(xpf, xpb, k * NFEAT);
    // h = relu(Ap @ (xp @ Wg) + bg)  [k,32]
    gemmTall(xpb, Wgb[i], zf, nullptr, k, NFILT, NFEAT, 0);
    cvt(zf, zb, k * NFILT);
    gemmTall(Apb[i], zb, hf, bg[i], k, NFILT, k, 2);
    // xagg[i] = BN(h)
    bn_cols<<<dim3(NFILT), dim3(256), 0, stream>>>(hf, gam[i], bet[i], zf, xab[i], k, NFILT);
  }

  // --- up path: only j=2 (k-index 0) reaches the output; au is factorized ---
  const int K0 = KS[0];
  // xu = S0 @ xagg0  [N,32]
  gemmTall(Sb[0], xab[0], zf, nullptr, NNODES, NFILT, K0, 0);
  cvt(zf, zb, NNODES * NFILT);
  // z = xu @ Wg5
  gemmTall(zb, Wgb[5], zf, nullptr, NNODES, NFILT, NFILT, 0);
  cvt(zf, zb, NNODES * NFILT);
  // t1 = S0^T z ; t2 = Ap0 t1 ; pre = S0 t2 + bg5, relu
  gemmTall(STb[0], zb, zf, nullptr, K0, NFILT, NNODES, 0);
  cvt(zf, zb, K0 * NFILT);
  gemmTall(Apb[0], zb, zf, nullptr, K0, NFILT, K0, 0);
  cvt(zf, zb, K0 * NFILT);
  gemmTall(Sb[0], zb, hf, bg[5], NNODES, NFILT, K0, 2);
  // xagg5 = BN
  bn_cols<<<dim3(NFILT), dim3(256), 0, stream>>>(hf, gam[5], bet[5], x5f, x5b, NNODES, NFILT);

  // --- final GCN: softmax(S0(Ap0(S0^T (xagg5 Wout))) + bout) ---
  gemmTall(x5b, Woutb, zf, nullptr, NNODES, NCLS, NFILT, 0);
  cvt(zf, zb, NNODES * NCLS);
  gemmTall(STb[0], zb, zf, nullptr, K0, NCLS, NNODES, 0);
  cvt(zf, zb, K0 * NCLS);
  gemmTall(Apb[0], zb, zf, nullptr, K0, NCLS, K0, 0);
  cvt(zf, zb, K0 * NCLS);
  gemmTall(Sb[0], zb, (float*)d_out, bout, NNODES, NCLS, K0, 1);
  softmax_rows<<<dim3(NNODES), dim3(256), 0, stream>>>((float*)d_out, NCLS);
}
